// MultiHeadAttention_15719580303928
// MI455X (gfx1250) — matrix operand order repaired and checked by reading, not compiled, not run on hardware
//
#include <hip/hip_runtime.h>
#include <cstdint>

// ---------------------------------------------------------------------------
// MI455X (gfx1250) multi-head causal attention, bf16 WMMA path.
// D=1024, H=16, Hd=64, B=2, S=2048.  Compute-bound (~100 GF vs ~50 MB HBM),
// so everything heavy runs through v_wmma_f32_16x16x32_bf16 (f32 accumulate).
// K-block staging uses CDNA5 async global->LDS DMA (ASYNCcnt).
// ---------------------------------------------------------------------------

typedef __bf16 bf16;
typedef bf16  v16bf  __attribute__((ext_vector_type(16)));
typedef bf16  bf16x8 __attribute__((ext_vector_type(8)));
typedef bf16  bf16x4 __attribute__((ext_vector_type(4)));
typedef float v8f    __attribute__((ext_vector_type(8)));

#define D_MODEL 1024
#define SEQ     2048
#define BATCH   2
#define NHEADS  16
#define HD      64
#define MTOT    (BATCH * SEQ)   // 4096

__device__ __forceinline__ v8f zero8() {
  v8f z = {0.f, 0.f, 0.f, 0.f, 0.f, 0.f, 0.f, 0.f};
  return z;
}

// A-matrix fragment, 16x32 bf16 (MxK), row-major source with leading dim ld.
__device__ __forceinline__ v16bf load_fragA(const bf16* base, int ld, int lane) {
  const int m  = lane & 15;
  const int kh = (lane >> 4) << 3;           // 0 or 8
  const bf16* p = base + (size_t)m * ld + kh;
  union { v16bf v; bf16x8 h[2]; } u;
  u.h[0] = *(const bf16x8*)(p);
  u.h[1] = *(const bf16x8*)(p + 16);
  return u.v;
}

// B-matrix fragment, 32x16 bf16 (KxN) where B[k][n] = src[n*ld + k].
__device__ __forceinline__ v16bf load_fragB(const bf16* base, int ld, int lane) {
  const int n  = lane & 15;
  const int k0 = (lane >> 4) << 3;           // 0 or 8
  const bf16* p = base + (size_t)n * ld + k0;
  union { v16bf v; bf16x8 h[2]; } u;
  u.h[0] = *(const bf16x8*)(p);
  u.h[1] = *(const bf16x8*)(p + 16);
  return u.v;
}

__device__ __forceinline__ v8f wmma_bf16(v16bf a, v16bf b, v8f c) {
  return __builtin_amdgcn_wmma_f32_16x16x32_bf16(
      /*neg_a=*/false, a, /*neg_b=*/false, b,
      /*c_mod=*/(short)0, c, /*reuse_a=*/false, /*reuse_b=*/false);
}

// ---------------------------------------------------------------------------
// fp32 -> bf16 conversion, vectorized (float4 in, bf16x4 out)
// ---------------------------------------------------------------------------
__global__ void cvt_f32_to_bf16(const float4* __restrict__ in,
                                bf16x4* __restrict__ out, int n4) {
  int i = blockIdx.x * blockDim.x + threadIdx.x;
  if (i < n4) {
    float4 f = in[i];
    bf16x4 o;
    o.x = (bf16)f.x; o.y = (bf16)f.y; o.z = (bf16)f.z; o.w = (bf16)f.w;
    out[i] = o;
  }
}

// ---------------------------------------------------------------------------
// C[m,n] = scale * sum_k A[m,k] * W[n,k]   (y = x @ W^T, W stored [out,in])
// Wave tile: 32(M) x 64(N). Block = 4 waves stacked in M -> 128x64 per WG.
// ---------------------------------------------------------------------------
template <bool F32OUT>
__global__ __launch_bounds__(128) void gemm_xwT(
    const bf16* __restrict__ A, const bf16* __restrict__ W,
    bf16* __restrict__ Cb, float* __restrict__ Cf,
    int M, int N, int K, float scale) {
  const int lane = threadIdx.x & 31;
  const int wv   = threadIdx.x >> 5;
  const int m0   = blockIdx.y * 128 + wv * 32;
  const int n0   = blockIdx.x * 64;

  v8f acc[2][4];
#pragma unroll
  for (int i = 0; i < 2; ++i)
#pragma unroll
    for (int j = 0; j < 4; ++j) acc[i][j] = zero8();

  for (int k0 = 0; k0 < K; k0 += 32) {
    if (k0 + 32 < K) {
      __builtin_prefetch(A + (size_t)m0 * K + k0 + 32, 0, 0);
      __builtin_prefetch(W + (size_t)n0 * K + k0 + 32, 0, 0);
    }
    v16bf a0 = load_fragA(A + (size_t)(m0 +  0) * K + k0, K, lane);
    v16bf a1 = load_fragA(A + (size_t)(m0 + 16) * K + k0, K, lane);
    v16bf b0 = load_fragB(W + (size_t)(n0 +  0) * K + k0, K, lane);
    v16bf b1 = load_fragB(W + (size_t)(n0 + 16) * K + k0, K, lane);
    v16bf b2 = load_fragB(W + (size_t)(n0 + 32) * K + k0, K, lane);
    v16bf b3 = load_fragB(W + (size_t)(n0 + 48) * K + k0, K, lane);
    acc[0][0] = wmma_bf16(a0, b0, acc[0][0]);
    acc[0][1] = wmma_bf16(a0, b1, acc[0][1]);
    acc[0][2] = wmma_bf16(a0, b2, acc[0][2]);
    acc[0][3] = wmma_bf16(a0, b3, acc[0][3]);
    acc[1][0] = wmma_bf16(a1, b0, acc[1][0]);
    acc[1][1] = wmma_bf16(a1, b1, acc[1][1]);
    acc[1][2] = wmma_bf16(a1, b2, acc[1][2]);
    acc[1][3] = wmma_bf16(a1, b3, acc[1][3]);
  }

  const int n  = lane & 15;
  const int hi = lane >> 4;
#pragma unroll
  for (int i = 0; i < 2; ++i)
#pragma unroll
    for (int j = 0; j < 4; ++j)
#pragma unroll
      for (int r = 0; r < 8; ++r) {
        const int m  = m0 + 16 * i + r + 8 * hi;
        const int nn = n0 + 16 * j + n;
        const float v = acc[i][j][r] * scale;
        if (F32OUT) Cf[(size_t)m * N + nn] = v;
        else        Cb[(size_t)m * N + nn] = (bf16)v;
      }
}

// ---------------------------------------------------------------------------
// Flash attention, causal. Block = 4 waves = 64 query rows; wave = 16 rows.
// 64-key blocks staged in LDS: K via async global->LDS DMA, V transposed via
// registers. 16 WMMA per key block between barriers.
// ---------------------------------------------------------------------------
__global__ __launch_bounds__(128) void flash_attn(
    const bf16* __restrict__ Qb, const bf16* __restrict__ Kb,
    const bf16* __restrict__ Vb, bf16* __restrict__ Ctx) {
  __shared__ bf16 sK[64 * 64];        // key block, row-major [k][d]   (8 KB)
  __shared__ bf16 sVt[64 * 64];       // value block transposed [d][k] (8 KB)
  __shared__ bf16 sP[4][16 * 64];     // per-wave probabilities [m][k] (8 KB)

  const int lane = threadIdx.x & 31;
  const int wv   = threadIdx.x >> 5;
  const int bh   = blockIdx.y;
  const int b    = bh >> 4;
  const int h    = bh & 15;
  const int q0   = blockIdx.x * 64;
  const int qw   = q0 + wv * 16;

  const size_t rowbase = (size_t)b * SEQ;
  const bf16* Qhead = Qb + rowbase * D_MODEL + h * HD;
  const bf16* Khead = Kb + rowbase * D_MODEL + h * HD;
  const bf16* Vhead = Vb + rowbase * D_MODEL + h * HD;

  // LDS byte offset of sK (low 32 bits of the generic address are the offset)
  const unsigned sK_base = (unsigned)(uintptr_t)(void*)&sK[0];

  // Q fragments for this wave's 16 rows (d = 0..31 and 32..63), loaded once.
  v16bf qf0 = load_fragA(Qhead + (size_t)qw * D_MODEL +  0, D_MODEL, lane);
  v16bf qf1 = load_fragA(Qhead + (size_t)qw * D_MODEL + 32, D_MODEL, lane);

  v8f o[4];
  float mi[8], li[8];
#pragma unroll
  for (int j = 0; j < 4; ++j) o[j] = zero8();
#pragma unroll
  for (int r = 0; r < 8; ++r) { mi[r] = -1e30f; li[r] = 0.f; }

  const int n  = lane & 15;
  const int hi = lane >> 4;
  const int nkb = blockIdx.x + 1;     // causal: key blocks with kb <= q0+63

  for (int ib = 0; ib < nkb; ++ib) {
    const int kb = ib * 64;

    // --- K block: async global->LDS DMA. 64 rows x 128 B: a layout-
    //     preserving copy, so each lane moves 16 B chunks. 512 chunks total.
#pragma unroll
    for (int it = 0; it < 4; ++it) {
      const int c  = (int)threadIdx.x + it * 128;
      const int kr = c >> 3;
      const int d0 = (c & 7) << 3;
      const unsigned lds = sK_base + (unsigned)c * 16u;
      const unsigned long long ga = (unsigned long long)(uintptr_t)
          (Khead + (size_t)(kb + kr) * D_MODEL + d0);
      asm volatile("global_load_async_to_lds_b128 %0, %1, off"
                   :: "v"(lds), "v"(ga) : "memory");
    }

    // --- V block: staged transposed ([d][k]) through registers.
    for (int c = threadIdx.x; c < 512; c += 128) {
      const int kr = c >> 3;
      const int d0 = (c & 7) << 3;
      bf16x8 vv = *(const bf16x8*)(Vhead + (size_t)(kb + kr) * D_MODEL + d0);
#pragma unroll
      for (int j = 0; j < 8; ++j) sVt[(d0 + j) * 64 + kr] = vv[j];
    }
    asm volatile("s_wait_asynccnt 0x0" ::: "memory");
    __syncthreads();

    // --- scores S = Q K^T : 4 tiles of 16x16 covering 64 keys.
    //     Batch all fragment loads so one dscnt wait covers the group.
    v16bf kf[8];
#pragma unroll
    for (int t = 0; t < 4; ++t) {
      kf[2 * t]     = load_fragB(&sK[(16 * t) * 64 +  0], 64, lane);
      kf[2 * t + 1] = load_fragB(&sK[(16 * t) * 64 + 32], 64, lane);
    }
    v8f s[4];
#pragma unroll
    for (int t = 0; t < 4; ++t) {
      v8f st = zero8();
      st = wmma_bf16(qf0, kf[2 * t],     st);
      st = wmma_bf16(qf1, kf[2 * t + 1], st);
      s[t] = st;
    }

    // --- causal mask (element (m,n): m = r + 8*hi, key = kb + 16t + n)
#pragma unroll
    for (int t = 0; t < 4; ++t)
#pragma unroll
      for (int r = 0; r < 8; ++r) {
        const int q = qw + r + 8 * hi;
        if (kb + 16 * t + n > q) s[t][r] = -1e30f;
      }

    // --- online softmax (row stats replicated across each 16-lane half)
    float mb[8];
#pragma unroll
    for (int r = 0; r < 8; ++r)
      mb[r] = fmaxf(fmaxf(s[0][r], s[1][r]), fmaxf(s[2][r], s[3][r]));
#pragma unroll
    for (int off = 1; off <= 8; off <<= 1)
#pragma unroll
      for (int r = 0; r < 8; ++r)
        mb[r] = fmaxf(mb[r], __shfl_xor(mb[r], off, 32));

    float al[8], rs[8];
#pragma unroll
    for (int r = 0; r < 8; ++r) {
      const float mn = fmaxf(mi[r], mb[r]);
      al[r] = __expf(mi[r] - mn);
      mi[r] = mn;
#pragma unroll
      for (int t = 0; t < 4; ++t) s[t][r] = __expf(s[t][r] - mn);
      rs[r] = (s[0][r] + s[1][r]) + (s[2][r] + s[3][r]);
    }
#pragma unroll
    for (int off = 1; off <= 8; off <<= 1)
#pragma unroll
      for (int r = 0; r < 8; ++r) rs[r] += __shfl_xor(rs[r], off, 32);
#pragma unroll
    for (int r = 0; r < 8; ++r) li[r] = li[r] * al[r] + rs[r];
#pragma unroll
    for (int j = 0; j < 4; ++j)
#pragma unroll
      for (int r = 0; r < 8; ++r) o[j][r] *= al[r];

    // --- C-layout -> A-layout relayout of P through per-wave LDS
    bf16* pw = sP[wv];
#pragma unroll
    for (int t = 0; t < 4; ++t)
#pragma unroll
      for (int r = 0; r < 8; ++r)
        pw[(r + 8 * hi) * 64 + 16 * t + n] = (bf16)s[t][r];

    v16bf pa0 = load_fragA(pw,      64, lane);   // keys 0..31
    v16bf pa1 = load_fragA(pw + 32, 64, lane);   // keys 32..63

    // --- O += P V  (B fragments contiguous thanks to transposed sVt)
    v16bf vf[8];
#pragma unroll
    for (int j = 0; j < 4; ++j) {
      vf[2 * j]     = load_fragB(&sVt[(16 * j) * 64 +  0], 64, lane);
      vf[2 * j + 1] = load_fragB(&sVt[(16 * j) * 64 + 32], 64, lane);
    }
#pragma unroll
    for (int j = 0; j < 4; ++j) {
      o[j] = wmma_bf16(pa0, vf[2 * j],     o[j]);
      o[j] = wmma_bf16(pa1, vf[2 * j + 1], o[j]);
    }

    __syncthreads();  // protect sK/sVt before next staging pass
  }

  // --- epilogue: O / l, write bf16 context in [B,S,D] layout
#pragma unroll
  for (int r = 0; r < 8; ++r) {
    const float inv = 1.0f / li[r];
    const int q = qw + r + 8 * hi;
    bf16* dst = Ctx + ((size_t)(b * SEQ + q)) * D_MODEL + h * HD + n;
#pragma unroll
    for (int j = 0; j < 4; ++j) dst[16 * j] = (bf16)(o[j][r] * inv);
  }
}

// ---------------------------------------------------------------------------
// Host-side orchestration
// ---------------------------------------------------------------------------
extern "C" void kernel_launch(void* const* d_in, const int* in_sizes, int n_in,
                              void* d_out, int out_size, void* d_ws, size_t ws_size,
                              hipStream_t stream) {
  const float* x  = (const float*)d_in[0];
  const float* Wq = (const float*)d_in[1];
  const float* Wk = (const float*)d_in[2];
  const float* Wv = (const float*)d_in[3];
  const float* Wo = (const float*)d_in[4];
  float* out = (float*)d_out;

  const size_t XE = (size_t)MTOT * D_MODEL;       // 4,194,304
  const size_t WE = (size_t)D_MODEL * D_MODEL;    // 1,048,576

  bf16* xb   = (bf16*)d_ws;
  bf16* wqb  = xb  + XE;
  bf16* wkb  = wqb + WE;
  bf16* wvb  = wkb + WE;
  bf16* wob  = wvb + WE;
  bf16* Qbuf = wob + WE;
  bf16* Kbuf = Qbuf + XE;
  bf16* Vbuf = Kbuf + XE;
  bf16* Ctxb = Vbuf + XE;

  auto cvt = [&](const float* src, bf16* dst, size_t nelem) {
    int n4 = (int)(nelem / 4);
    cvt_f32_to_bf16<<<(n4 + 255) / 256, 256, 0, stream>>>(
        (const float4*)src, (bf16x4*)dst, n4);
  };
  cvt(x,  xb,  XE);
  cvt(Wq, wqb, WE);
  cvt(Wk, wkb, WE);
  cvt(Wv, wvb, WE);
  cvt(Wo, wob, WE);

  dim3 gg(D_MODEL / 64, MTOT / 128);   // (16, 32)
  // Q gets the 1/sqrt(Hd)=0.125 scale folded in (exact in bf16).
  gemm_xwT<false><<<gg, 128, 0, stream>>>(xb, wqb, Qbuf, (float*)nullptr,
                                          MTOT, D_MODEL, D_MODEL, 0.125f);
  gemm_xwT<false><<<gg, 128, 0, stream>>>(xb, wkb, Kbuf, (float*)nullptr,
                                          MTOT, D_MODEL, D_MODEL, 1.0f);
  gemm_xwT<false><<<gg, 128, 0, stream>>>(xb, wvb, Vbuf, (float*)nullptr,
                                          MTOT, D_MODEL, D_MODEL, 1.0f);

  flash_attn<<<dim3(SEQ / 64, BATCH * NHEADS), 128, 0, stream>>>(
      Qbuf, Kbuf, Vbuf, Ctxb);

  gemm_xwT<true><<<gg, 128, 0, stream>>>(Ctxb, wob, (bf16*)nullptr, out,
                                         MTOT, D_MODEL, D_MODEL, 1.0f);
}
